// GATConv_42417097016495
// MI455X (gfx1250) — matrix operand order repaired and checked by reading, not compiled, not run on hardware
//
#include <hip/hip_runtime.h>
#include <hip/hip_bf16.h>

typedef __attribute__((ext_vector_type(16))) _Float16 v16h;
typedef __attribute__((ext_vector_type(8)))  float    v8f;

#define INCH   64
#define OUTC   16
#define NHEAD  8
#define HC     128   // NHEAD*OUTC
#define NEG_SLOPE 0.2f

// ---------------------------------------------------------------------------
// Kernel 1: init out=0, amax=-inf, denom=0
// ---------------------------------------------------------------------------
__global__ void gat_init(float* __restrict__ out, float* __restrict__ amax,
                         float* __restrict__ denom, int n128, int n8) {
    int i = blockIdx.x * blockDim.x + threadIdx.x;
    if (i < n128) out[i] = 0.0f;
    if (i < n8) {
        amax[i]  = -__builtin_inff();
        denom[i] = 0.0f;
    }
}

// ---------------------------------------------------------------------------
// Kernel 2: h = x @ W via V_WMMA_F32_16X16X32_F16 (f32 accumulate),
//   fused per-head attention logits.
//   block = 256 threads = 8 waves; wave w owns output-column tile (= head) w.
//   Each block handles 16 nodes (one M tile); K=64 -> 2 WMMA k-steps.
// ---------------------------------------------------------------------------
__global__ __launch_bounds__(256) void gat_gemm(
    const float* __restrict__ x, const float* __restrict__ W,
    const float* __restrict__ att_src, const float* __restrict__ att_dst,
    float* __restrict__ h, float* __restrict__ asrc, float* __restrict__ adst,
    int N)
{
    __shared__ float xs[16 * 65];    // 16 rows x 64, padded to 65 floats
    __shared__ float Ws[64 * 130];   // 64 rows x 128, padded to 130 floats

    const int tid = threadIdx.x;
    const int m0  = blockIdx.x * 16;

    // cooperative load of W (64x128) into LDS
    for (int i = tid; i < 64 * 128; i += 256) {
        int k = i >> 7, n = i & 127;
        Ws[k * 130 + n] = W[i];
    }
    // cooperative load of x tile (16x64) into LDS (guard tail rows)
    for (int i = tid; i < 16 * 64; i += 256) {
        int m = i >> 6, k = i & 63;
        int row = m0 + m;
        xs[m * 65 + k] = (row < N) ? x[(size_t)row * INCH + k] : 0.0f;
    }
    __syncthreads();

    const int lane = tid & 31;
    const int wv   = tid >> 5;        // wave id == head id == N-tile id
    const int g    = lane >> 4;       // lane group 0/1
    const int l16  = lane & 15;
    const int n0   = wv * OUTC;

    v8f acc = {};
    #pragma unroll
    for (int k0 = 0; k0 < INCH; k0 += 32) {
        v16h a, b;
        // A 16x32 f16: row m=l16; VGPR j<4 -> K=8g+2j(+1); j>=4 -> K=16+8g+2(j-4)(+1)
        #pragma unroll
        for (int j = 0; j < 8; ++j) {
            int base = k0 + ((j < 4) ? 0 : 16) + 8 * g + 2 * (j & 3);
            a[2 * j]     = (_Float16)xs[l16 * 65 + base];
            a[2 * j + 1] = (_Float16)xs[l16 * 65 + base + 1];
        }
        // B 32x16 f16: col n=n0+l16; VGPR j<4 -> K=8g+2j(+1); j>=4 -> K=16+8g+2(j-4)(+1)
        #pragma unroll
        for (int j = 0; j < 8; ++j) {
            int K = k0 + ((j < 4) ? 0 : 16) + 8 * g + 2 * (j & 3);
            b[2 * j]     = (_Float16)Ws[K * 130 + n0 + l16];
            b[2 * j + 1] = (_Float16)Ws[(K + 1) * 130 + n0 + l16];
        }
        acc = __builtin_amdgcn_wmma_f32_16x16x32_f16(
            false, a, false, b, (short)0, acc, false, false);
    }

    // D 16x16 f32: VGPR i -> row i (lanes 0-15) / row i+8 (lanes 16-31)
    #pragma unroll
    for (int i = 0; i < 8; ++i) {
        int row = m0 + i + 8 * g;
        if (row < N) h[(size_t)row * HC + n0 + l16] = acc[i];
    }

    // fused per-head attention logits: a_src[n,wv] = sum_c h[n,wv,c]*att_src[wv,c]
    const float av = att_src[wv * OUTC + l16];
    const float dv = att_dst[wv * OUTC + l16];
    #pragma unroll
    for (int i = 0; i < 8; ++i) {
        float ss = acc[i] * av;
        float sd = acc[i] * dv;
        #pragma unroll
        for (int off = 1; off < 16; off <<= 1) {
            ss += __shfl_xor(ss, off, 16);
            sd += __shfl_xor(sd, off, 16);
        }
        if (l16 == 0) {
            int row = m0 + i + 8 * g;
            if (row < N) {
                asrc[(size_t)row * NHEAD + wv] = ss;
                adst[(size_t)row * NHEAD + wv] = sd;
            }
        }
    }
}

// ---------------------------------------------------------------------------
// Edge helpers
// ---------------------------------------------------------------------------
__device__ __forceinline__ void atomic_max_f32(float* addr, float val) {
    // exact float max via sign-split integer atomics (init must be -inf)
    if (val >= 0.0f)
        atomicMax((int*)addr, __float_as_int(val));
    else
        atomicMin((unsigned int*)addr, (unsigned int)__float_as_int(val));
}

__device__ __forceinline__ void edge_endpoints(const int* __restrict__ ei,
                                               int e, int E, int& s, int& d) {
    if (e < E) { s = ei[e]; d = ei[E + e]; }
    else       { s = d = e - E; }            // self loops appended
}

// ---------------------------------------------------------------------------
// Kernel 3: segment max of leaky_relu logits over destination
//   one thread per (edge, head)
// ---------------------------------------------------------------------------
__global__ void gat_edge_max(const int* __restrict__ ei,
                             const float* __restrict__ asrc,
                             const float* __restrict__ adst,
                             float* __restrict__ amax, int E, int total) {
    int gid = blockIdx.x * blockDim.x + threadIdx.x;
    int e = gid >> 3, hd = gid & 7;
    if (e >= total) return;
    int s, d; edge_endpoints(ei, e, E, s, d);
    float a = asrc[(size_t)s * NHEAD + hd] + adst[(size_t)d * NHEAD + hd];
    a = (a > 0.0f) ? a : NEG_SLOPE * a;
    atomic_max_f32(&amax[(size_t)d * NHEAD + hd], a);
}

// ---------------------------------------------------------------------------
// Kernel 4: segment sum of exp(alpha - amax) over destination
// ---------------------------------------------------------------------------
__global__ void gat_edge_sum(const int* __restrict__ ei,
                             const float* __restrict__ asrc,
                             const float* __restrict__ adst,
                             const float* __restrict__ amax,
                             float* __restrict__ denom, int E, int total) {
    int gid = blockIdx.x * blockDim.x + threadIdx.x;
    int e = gid >> 3, hd = gid & 7;
    if (e >= total) return;
    int s, d; edge_endpoints(ei, e, E, s, d);
    float a = asrc[(size_t)s * NHEAD + hd] + adst[(size_t)d * NHEAD + hd];
    a = (a > 0.0f) ? a : NEG_SLOPE * a;
    float ex = expf(a - amax[(size_t)d * NHEAD + hd]);
    atomicAdd(&denom[(size_t)d * NHEAD + hd], ex);
}

// ---------------------------------------------------------------------------
// Kernel 5: weighted message scatter: out[dst] += attn * h[src]
//   one thread per (edge, head), 16 channels each (4x float4 gather)
// ---------------------------------------------------------------------------
__global__ void gat_scatter(const int* __restrict__ ei,
                            const float* __restrict__ asrc,
                            const float* __restrict__ adst,
                            const float* __restrict__ amax,
                            const float* __restrict__ denom,
                            const float* __restrict__ h,
                            float* __restrict__ out, int E, int total) {
    int gid = blockIdx.x * blockDim.x + threadIdx.x;
    int e = gid >> 3, hd = gid & 7;
    if (e >= total) return;
    int s, d; edge_endpoints(ei, e, E, s, d);
    float a = asrc[(size_t)s * NHEAD + hd] + adst[(size_t)d * NHEAD + hd];
    a = (a > 0.0f) ? a : NEG_SLOPE * a;
    float attn = expf(a - amax[(size_t)d * NHEAD + hd]) /
                 (denom[(size_t)d * NHEAD + hd] + 1e-16f);

    const float4* hp = (const float4*)(h + (size_t)s * HC + hd * OUTC);
    float* op = out + (size_t)d * HC + hd * OUTC;
    #pragma unroll
    for (int q = 0; q < 4; ++q) {
        float4 v = hp[q];
        atomicAdd(op + 4 * q + 0, v.x * attn);
        atomicAdd(op + 4 * q + 1, v.y * attn);
        atomicAdd(op + 4 * q + 2, v.z * attn);
        atomicAdd(op + 4 * q + 3, v.w * attn);
    }
}

// ---------------------------------------------------------------------------
// Kernel 6: out = relu(out + bias)
// ---------------------------------------------------------------------------
__global__ void gat_final(float* __restrict__ out, const float* __restrict__ bias,
                          int n128) {
    int i = blockIdx.x * blockDim.x + threadIdx.x;
    if (i < n128) {
        float v = out[i] + bias[i & (HC - 1)];
        out[i] = v > 0.0f ? v : 0.0f;
    }
}

// ---------------------------------------------------------------------------
extern "C" void kernel_launch(void* const* d_in, const int* in_sizes, int n_in,
                              void* d_out, int out_size, void* d_ws, size_t ws_size,
                              hipStream_t stream) {
    const float* x       = (const float*)d_in[0];
    const float* W       = (const float*)d_in[1];
    const float* att_src = (const float*)d_in[2];
    const float* att_dst = (const float*)d_in[3];
    const float* bias    = (const float*)d_in[4];
    const int*   ei      = (const int*)d_in[5];

    const int N = in_sizes[0] / INCH;
    const int E = in_sizes[5] / 2;

    float* out   = (float*)d_out;
    float* ws    = (float*)d_ws;
    float* h     = ws;                               // N*128
    float* asrc  = h    + (size_t)N * HC;            // N*8
    float* adst  = asrc + (size_t)N * NHEAD;         // N*8
    float* amax  = adst + (size_t)N * NHEAD;         // N*8
    float* denom = amax + (size_t)N * NHEAD;         // N*8

    const int n128 = N * HC;
    const int n8   = N * NHEAD;
    const int total = E + N;               // edges + self loops
    const int et    = total * NHEAD;       // (edge, head) threads

    gat_init<<<(n128 + 255) / 256, 256, 0, stream>>>(out, amax, denom, n128, n8);
    gat_gemm<<<(N + 15) / 16, 256, 0, stream>>>(x, W, att_src, att_dst,
                                                h, asrc, adst, N);
    gat_edge_max<<<(et + 255) / 256, 256, 0, stream>>>(ei, asrc, adst, amax, E, total);
    gat_edge_sum<<<(et + 255) / 256, 256, 0, stream>>>(ei, asrc, adst, amax, denom, E, total);
    gat_scatter<<<(et + 255) / 256, 256, 0, stream>>>(ei, asrc, adst, amax, denom,
                                                      h, out, E, total);
    gat_final<<<(n128 + 255) / 256, 256, 0, stream>>>(out, bias, n128);
}
